// GCNBaseline_44160853737915
// MI455X (gfx1250) — compile-verified, matrix operand order checked by reading, not run on hardware
//
#include <hip/hip_runtime.h>
#include <hip/hip_bf16.h>

typedef _Float16 f16;
typedef __attribute__((ext_vector_type(16))) _Float16 v16h;
typedef __attribute__((ext_vector_type(8)))  _Float16 v8h;
typedef __attribute__((ext_vector_type(8)))  float    v8f;

__device__ __forceinline__ v16h gcn_cat8(v8h lo, v8h hi) {
  return __builtin_shufflevector(lo, hi, 0,1,2,3,4,5,6,7,8,9,10,11,12,13,14,15);
}

// ---------------- elementwise / graph kernels ----------------

__global__ void gcn_fill_f32(float* __restrict__ p, float v, int n) {
  int i = blockIdx.x * blockDim.x + threadIdx.x;
  if (i < n) p[i] = v;
}

__global__ void gcn_deg_accum(const int* __restrict__ dst, float* __restrict__ deg, int e) {
  int i = blockIdx.x * blockDim.x + threadIdx.x;
  if (i < e) atomicAdd(&deg[dst[i]], 1.0f);
}

__global__ void gcn_rsqrt(const float* __restrict__ deg, float* __restrict__ dinv, int n) {
  int i = blockIdx.x * blockDim.x + threadIdx.x;
  if (i < n) { float d = deg[i]; dinv[i] = d > 0.f ? rsqrtf(d) : 0.f; }
}

__global__ void gcn_cvt_f16(const float* __restrict__ s, f16* __restrict__ d, size_t n) {
  size_t i = (size_t)blockIdx.x * blockDim.x + threadIdx.x;
  if (i < n) d[i] = (f16)s[i];
}

// W is [K,N] row-major; write Bt[n*K + k] (column-major) so B loads match A's pattern.
__global__ void gcn_cvtT_f16(const float* __restrict__ W, f16* __restrict__ Bt, int K, int N) {
  int i = blockIdx.x * blockDim.x + threadIdx.x;
  if (i >= K * N) return;
  int k = i / N, n = i - k * N;
  Bt[(size_t)n * K + k] = (f16)W[i];
}

// agg[i,c] = h[i,c] * dinv[i]^2   (self-loop term, also zero-initializes agg)
__global__ void gcn_selfinit(const float* __restrict__ h, const float* __restrict__ dinv,
                             float* __restrict__ agg, size_t total, int cshift) {
  size_t i = (size_t)blockIdx.x * blockDim.x + threadIdx.x;
  if (i < total) { float di = dinv[i >> cshift]; agg[i] = h[i] * di * di; }
}

// One thread per (edge, 4 channels): agg[dst] += h[src] * dinv[src]*dinv[dst]
// agg is L2-resident (<= 51 MB) so the f32 atomics resolve at L2, not HBM.
__global__ void gcn_scatter(const float* __restrict__ h, const int* __restrict__ src,
                            const int* __restrict__ dst, const float* __restrict__ dinv,
                            float* __restrict__ agg, int E, int cshift /*log2(C/4)*/, int C) {
  size_t idx = (size_t)blockIdx.x * blockDim.x + threadIdx.x;
  int e = (int)(idx >> cshift);
  if (e >= E) return;
  int g = ((int)idx & ((1 << cshift) - 1)) << 2;
  int s = src[e], d = dst[e];
  float coef = dinv[s] * dinv[d];
  const float4 v = *(const float4*)(h + (size_t)s * C + g);
  float* o = agg + (size_t)d * C + g;
  atomicAdd(o + 0, v.x * coef);
  atomicAdd(o + 1, v.y * coef);
  atomicAdd(o + 2, v.z * coef);
  atomicAdd(o + 3, v.w * coef);
}

__global__ void gcn_bias_relu_cvt(const float* __restrict__ agg, const float* __restrict__ b,
                                  f16* __restrict__ out, size_t total, int cmask) {
  size_t i = (size_t)blockIdx.x * blockDim.x + threadIdx.x;
  if (i < total) { float v = agg[i] + b[(int)(i & (size_t)cmask)]; out[i] = (f16)fmaxf(v, 0.f); }
}

__global__ void gcn_bias_inplace(float* __restrict__ agg, const float* __restrict__ b,
                                 size_t total, int cmask) {
  size_t i = (size_t)blockIdx.x * blockDim.x + threadIdx.x;
  if (i < total) agg[i] += b[(int)(i & (size_t)cmask)];
}

// pf[p] = [zs, zd, zs*zd, |zs-zd|] in f16 ; EMB_C = 64, PF_C = 256
__global__ void gcn_pairfeat(const float* __restrict__ z, const int* __restrict__ eli,
                             f16* __restrict__ pf, int P) {
  size_t idx = (size_t)blockIdx.x * blockDim.x + threadIdx.x;
  if (idx >= (size_t)P * 64) return;
  int p = (int)(idx >> 6), c = (int)(idx & 63);
  int s = eli[p], d = eli[P + p];
  float zs = z[(size_t)s * 64 + c];
  float zd = z[(size_t)d * 64 + c];
  f16* row = pf + (size_t)p * 256;
  row[c]        = (f16)zs;
  row[64 + c]   = (f16)zd;
  row[128 + c]  = (f16)(zs * zd);
  row[192 + c]  = (f16)fabsf(zs - zd);
}

__global__ void gcn_init_out(float* __restrict__ out, const float* __restrict__ bm2, int P) {
  int i = blockIdx.x * blockDim.x + threadIdx.x;
  if (i < P) out[i] = bm2[0];
}

// ---------------- WMMA GEMM kernels (register-blocked: 16x64 of C per wave) ----------
// A: [M,K] f16 row-major ; Bt: [N,K] f16 (column-major B) ; C: [M,N] f32 row-major.
// Each wave computes one M-tile x 4 N-tiles: per K-step it loads 1 A-frag + 4 B-frags
// and issues 4 WMMAs, quartering A traffic vs one-tile-per-wave.
// 16-bit operand ISA layout (16x32): lane l(<16): row/col = l, elems 0..7 = K kb..kb+7,
// elems 8..15 = K kb+16..kb+23, where kb = k0 + 8*(lane>=16).
__global__ void gcn_gemm_f16(const f16* __restrict__ A, const f16* __restrict__ Bt,
                             float* __restrict__ C, int mtiles, int N, int K, int ngshift) {
  int wave = blockIdx.x * (blockDim.x >> 5) + (threadIdx.x >> 5);
  int mt = wave >> ngshift;
  if (mt >= mtiles) return;                 // wave-uniform: EXEC stays all-ones at WMMA
  int ng = wave & ((1 << ngshift) - 1);     // group of 4 N-tiles (64 cols)
  int lane = threadIdx.x & 31;
  int l = lane & 15, half = lane >> 4;
  const f16* Ar = A + (size_t)(mt * 16 + l) * K;
  const f16* Br0 = Bt + (size_t)(ng * 64 + 0 * 16 + l) * K;
  const f16* Br1 = Bt + (size_t)(ng * 64 + 1 * 16 + l) * K;
  const f16* Br2 = Bt + (size_t)(ng * 64 + 2 * 16 + l) * K;
  const f16* Br3 = Bt + (size_t)(ng * 64 + 3 * 16 + l) * K;
  v8f acc0 = {}, acc1 = {}, acc2 = {}, acc3 = {};
  for (int k0 = 0; k0 < K; k0 += 32) {
    int kb = k0 + half * 8;
    v16h a  = gcn_cat8(*(const v8h*)(Ar  + kb), *(const v8h*)(Ar  + kb + 16));
    v16h b0 = gcn_cat8(*(const v8h*)(Br0 + kb), *(const v8h*)(Br0 + kb + 16));
    v16h b1 = gcn_cat8(*(const v8h*)(Br1 + kb), *(const v8h*)(Br1 + kb + 16));
    v16h b2 = gcn_cat8(*(const v8h*)(Br2 + kb), *(const v8h*)(Br2 + kb + 16));
    v16h b3 = gcn_cat8(*(const v8h*)(Br3 + kb), *(const v8h*)(Br3 + kb + 16));
    acc0 = __builtin_amdgcn_wmma_f32_16x16x32_f16(false, a, false, b0, (short)0, acc0, false, false);
    acc1 = __builtin_amdgcn_wmma_f32_16x16x32_f16(false, a, false, b1, (short)0, acc1, false, false);
    acc2 = __builtin_amdgcn_wmma_f32_16x16x32_f16(false, a, false, b2, (short)0, acc2, false, false);
    acc3 = __builtin_amdgcn_wmma_f32_16x16x32_f16(false, a, false, b3, (short)0, acc3, false, false);
  }
  // C/D layout: VGPR r -> row mt*16 + 8*half + r ; lane%16 -> col within tile.
  int row0 = mt * 16 + half * 8;
  int colb = ng * 64 + l;
  float* Cr = C + (size_t)row0 * N + colb;
#pragma unroll
  for (int r = 0; r < 8; ++r) {
    Cr[0]  = acc0[r];
    Cr[16] = acc1[r];
    Cr[32] = acc2[r];
    Cr[48] = acc3[r];
    Cr += N;
  }
}

// Same blocked GEMM with fused head epilogue: v = sum_j relu(acc_j + bm1_j) * Wm2_j over
// the wave's 64 columns, one cross-lane reduce per row, atomicAdd into logits
// (pre-initialized to bm2). Only N/64 atomics per logit.
__global__ void gcn_gemm_head_f16(const f16* __restrict__ A, const f16* __restrict__ Bt,
                                  const float* __restrict__ bias, const float* __restrict__ w2,
                                  float* __restrict__ out, int mtiles, int K, int ngshift) {
  int wave = blockIdx.x * (blockDim.x >> 5) + (threadIdx.x >> 5);
  int mt = wave >> ngshift;
  if (mt >= mtiles) return;
  int ng = wave & ((1 << ngshift) - 1);
  int lane = threadIdx.x & 31;
  int l = lane & 15, half = lane >> 4;
  const f16* Ar = A + (size_t)(mt * 16 + l) * K;
  const f16* Br0 = Bt + (size_t)(ng * 64 + 0 * 16 + l) * K;
  const f16* Br1 = Bt + (size_t)(ng * 64 + 1 * 16 + l) * K;
  const f16* Br2 = Bt + (size_t)(ng * 64 + 2 * 16 + l) * K;
  const f16* Br3 = Bt + (size_t)(ng * 64 + 3 * 16 + l) * K;
  v8f acc0 = {}, acc1 = {}, acc2 = {}, acc3 = {};
  for (int k0 = 0; k0 < K; k0 += 32) {
    int kb = k0 + half * 8;
    v16h a  = gcn_cat8(*(const v8h*)(Ar  + kb), *(const v8h*)(Ar  + kb + 16));
    v16h b0 = gcn_cat8(*(const v8h*)(Br0 + kb), *(const v8h*)(Br0 + kb + 16));
    v16h b1 = gcn_cat8(*(const v8h*)(Br1 + kb), *(const v8h*)(Br1 + kb + 16));
    v16h b2 = gcn_cat8(*(const v8h*)(Br2 + kb), *(const v8h*)(Br2 + kb + 16));
    v16h b3 = gcn_cat8(*(const v8h*)(Br3 + kb), *(const v8h*)(Br3 + kb + 16));
    acc0 = __builtin_amdgcn_wmma_f32_16x16x32_f16(false, a, false, b0, (short)0, acc0, false, false);
    acc1 = __builtin_amdgcn_wmma_f32_16x16x32_f16(false, a, false, b1, (short)0, acc1, false, false);
    acc2 = __builtin_amdgcn_wmma_f32_16x16x32_f16(false, a, false, b2, (short)0, acc2, false, false);
    acc3 = __builtin_amdgcn_wmma_f32_16x16x32_f16(false, a, false, b3, (short)0, acc3, false, false);
  }
  int row0 = mt * 16 + half * 8;
  int colb = ng * 64 + l;
  float bc0 = bias[colb], bc1 = bias[colb + 16], bc2 = bias[colb + 32], bc3 = bias[colb + 48];
  float wc0 = w2[colb],   wc1 = w2[colb + 16],   wc2 = w2[colb + 32],   wc3 = w2[colb + 48];
#pragma unroll
  for (int r = 0; r < 8; ++r) {
    float v = fmaxf(acc0[r] + bc0, 0.f) * wc0
            + fmaxf(acc1[r] + bc1, 0.f) * wc1
            + fmaxf(acc2[r] + bc2, 0.f) * wc2
            + fmaxf(acc3[r] + bc3, 0.f) * wc3;
    v += __shfl_xor(v, 1, 16);
    v += __shfl_xor(v, 2, 16);
    v += __shfl_xor(v, 4, 16);
    v += __shfl_xor(v, 8, 16);
    if (l == 0) atomicAdd(&out[row0 + r], v);
  }
}

// ---------------- launcher ----------------

extern "C" void kernel_launch(void* const* d_in, const int* in_sizes, int n_in,
                              void* d_out, int out_size, void* d_ws, size_t ws_size,
                              hipStream_t stream) {
  const int IN_C = 128, HID_C = 128, EMB_C = 64, PF_C = 256, PRED_H = 256;

  const float* x   = (const float*)d_in[0];
  const int*   ei  = (const int*)d_in[1];   // [2,E]: row0=src, row1=dst
  const int*   eli = (const int*)d_in[2];   // [2,P]
  const float* W1  = (const float*)d_in[3];
  const float* b1  = (const float*)d_in[4];
  const float* W2  = (const float*)d_in[5];
  const float* b2  = (const float*)d_in[6];
  const float* Wm1 = (const float*)d_in[7];
  const float* bm1 = (const float*)d_in[8];
  const float* Wm2 = (const float*)d_in[9];  // [256,1] -> vector
  const float* bm2 = (const float*)d_in[10];
  float* out = (float*)d_out;

  const int Nn = in_sizes[0] / IN_C;
  const int E  = in_sizes[1] / 2;
  const int P  = in_sizes[2] / 2;

  // workspace carve (256B aligned)
  char* ws = (char*)d_ws;
  size_t off = 0;
  auto carve = [&](size_t bytes) -> char* {
    char* p = ws + off;
    off += (bytes + 255) & ~(size_t)255;
    return p;
  };
  float* deg  = (float*)carve((size_t)Nn * 4);
  float* dinv = (float*)carve((size_t)Nn * 4);
  f16*   xh   = (f16*)carve((size_t)Nn * IN_C * 2);
  f16*   W1t  = (f16*)carve((size_t)IN_C * HID_C * 2);
  f16*   W2t  = (f16*)carve((size_t)HID_C * EMB_C * 2);
  f16*   Wm1t = (f16*)carve((size_t)PF_C * PRED_H * 2);
  float* hlin = (float*)carve((size_t)Nn * HID_C * 4);   // reused as z_lin
  float* agg  = (float*)carve((size_t)Nn * HID_C * 4);   // reused as agg2/z
  f16*   h1h  = (f16*)carve((size_t)Nn * HID_C * 2);
  f16*   pf   = (f16*)carve((size_t)P * PF_C * 2);

  const int T = 256;
  auto blk = [](long long n, int t) { return (int)((n + t - 1) / t); };

  // --- degree / normalization (self-loop contributes +1 via init) ---
  gcn_fill_f32<<<blk(Nn, T), T, 0, stream>>>(deg, 1.0f, Nn);
  gcn_deg_accum<<<blk(E, T), T, 0, stream>>>(ei + E, deg, E);
  gcn_rsqrt<<<blk(Nn, T), T, 0, stream>>>(deg, dinv, Nn);

  // --- f16 conversions (weights transposed to column-major) ---
  gcn_cvt_f16<<<blk((long long)Nn * IN_C, T), T, 0, stream>>>(x, xh, (size_t)Nn * IN_C);
  gcn_cvtT_f16<<<blk(IN_C * HID_C, T), T, 0, stream>>>(W1, W1t, IN_C, HID_C);
  gcn_cvtT_f16<<<blk(HID_C * EMB_C, T), T, 0, stream>>>(W2, W2t, HID_C, EMB_C);
  gcn_cvtT_f16<<<blk(PF_C * PRED_H, T), T, 0, stream>>>(Wm1, Wm1t, PF_C, PRED_H);

  // --- conv1: h_lin = x @ W1 (WMMA, N=128 -> 2 N-groups) ---
  {
    int mtiles = (Nn + 15) / 16, ngshift = 1;
    int waves = mtiles << ngshift;
    gcn_gemm_f16<<<(waves + 3) / 4, 128, 0, stream>>>(xh, W1t, hlin, mtiles, HID_C, IN_C, ngshift);
  }
  gcn_selfinit<<<blk((long long)Nn * HID_C, T), T, 0, stream>>>(hlin, dinv, agg, (size_t)Nn * HID_C, 7);
  gcn_scatter<<<blk((long long)E * (HID_C / 4), T), T, 0, stream>>>(hlin, ei, ei + E, dinv, agg, E, 5, HID_C);
  gcn_bias_relu_cvt<<<blk((long long)Nn * HID_C, T), T, 0, stream>>>(agg, b1, h1h, (size_t)Nn * HID_C, HID_C - 1);

  // --- conv2: z_lin = h1 @ W2 (WMMA, N=64 -> 1 N-group) ---
  {
    int mtiles = (Nn + 15) / 16, ngshift = 0;
    int waves = mtiles << ngshift;
    gcn_gemm_f16<<<(waves + 3) / 4, 128, 0, stream>>>(h1h, W2t, hlin, mtiles, EMB_C, HID_C, ngshift);
  }
  gcn_selfinit<<<blk((long long)Nn * EMB_C, T), T, 0, stream>>>(hlin, dinv, agg, (size_t)Nn * EMB_C, 6);
  gcn_scatter<<<blk((long long)E * (EMB_C / 4), T), T, 0, stream>>>(hlin, ei, ei + E, dinv, agg, E, 4, EMB_C);
  gcn_bias_inplace<<<blk((long long)Nn * EMB_C, T), T, 0, stream>>>(agg, b2, (size_t)Nn * EMB_C, EMB_C - 1);

  // --- pair features + fused MLP head (N=256 -> 4 N-groups) ---
  gcn_pairfeat<<<blk((long long)P * EMB_C, T), T, 0, stream>>>(agg, eli, pf, P);
  gcn_init_out<<<blk(P, T), T, 0, stream>>>(out, bm2, P);
  {
    int mtiles = (P + 15) / 16, ngshift = 2;
    int waves = mtiles << ngshift;
    gcn_gemm_head_f16<<<(waves + 3) / 4, 128, 0, stream>>>(pf, Wm1t, bm1, Wm2, out, mtiles, PF_C, ngshift);
  }
}
